// MultiHeadSelfAttention_3341484556968
// MI455X (gfx1250) — matrix operand order repaired and checked by reading, not compiled, not run on hardware
//
#include <hip/hip_runtime.h>

#define DEV __device__ __forceinline__

typedef __attribute__((ext_vector_type(16))) __bf16       bf16x16;
typedef __attribute__((ext_vector_type(8)))  float        f32x8;
typedef __attribute__((ext_vector_type(4)))  unsigned int u32x4;

union FragU { u32x4 q[2]; unsigned dw[8]; bf16x16 v; };

static constexpr int Dm    = 1024;       // model dim
static constexpr int Hh    = 16;         // heads
static constexpr int DEP   = 64;         // head depth
static constexpr int Sq    = 2048;       // sequence
static constexpr int Bb    = 2;          // batch
static constexpr int Mrows = Bb * Sq;    // 4096 token rows
static constexpr int KT    = 32;         // keys per flash iteration
static constexpr int NKT   = Sq / KT;    // 64 key tiles

// ---------- helpers ----------
DEV unsigned short bf16_raw(float f) {           // fp32 -> bf16 (RNE)
  unsigned u = __builtin_bit_cast(unsigned, f);
  u += 0x7FFFu + ((u >> 16) & 1u);
  return (unsigned short)(u >> 16);
}
DEV unsigned pack2(float lo, float hi) {
  return (unsigned)bf16_raw(lo) | ((unsigned)bf16_raw(hi) << 16);
}

// lane <-> lane^16 exchange: v_permlanex16_b32 with identity selects
DEV unsigned swap16u(unsigned x) {
  return (unsigned)__builtin_amdgcn_permlanex16((int)x, (int)x, 0x76543210,
                                                0xfedcba98, false, false);
}
DEV float swap16f(float x) {
  return __builtin_bit_cast(float, (int)swap16u(__builtin_bit_cast(unsigned, x)));
}

// A-fragment (16x32 bf16, MxK) from global, row-major, K contiguous:
// lane<16 -> row=lane, K {0..7,16..23}; lane>=16 -> row=lane-16, K {8..15,24..31}
DEV bf16x16 load_fragA(const unsigned short* base, int ld) {
  const int lane = threadIdx.x & 31;
  const int r    = lane & 15;
  const int kb   = (lane >> 4) << 3;              // 0 or 8
  const unsigned short* p = base + (size_t)r * ld + kb;
  FragU f;
  f.q[0] = *(const u32x4*)(p);                    // K = kb .. kb+7
  f.q[1] = *(const u32x4*)(p + 16);               // K = kb+16 .. kb+23
  return f.v;
}

// B-fragment (32x16 bf16, KxN): lane<16 -> col=lane, K {0..7,16..23};
// lane>=16 -> col=lane-16, K {8..15,24..31}. Source = column-major-as-rows (TN GEMM).
DEV bf16x16 load_fragB(const unsigned short* base, int ld) {
  const int lane = threadIdx.x & 31;
  const int n    = lane & 15;
  const int ks   = (lane >> 4) << 3;              // 0 or 8
  const unsigned short* p = base + (size_t)n * ld + ks;
  FragU f;
  f.q[0] = *(const u32x4*)(p);                    // K = ks .. ks+7
  f.q[1] = *(const u32x4*)(p + 16);               // K = ks+16 .. ks+23
  return f.v;
}

DEV f32x8 wmma_bf16(bf16x16 a, bf16x16 b, f32x8 c) {
  return __builtin_amdgcn_wmma_f32_16x16x32_bf16(false, a, false, b,
                                                 (short)0, c, false, false);
}

// CDNA5 async global->LDS DMA (16B per lane), tracked by ASYNCcnt
DEV void async_ld16(unsigned lds_byte_addr, const unsigned short* gptr) {
  asm volatile("global_load_async_to_lds_b128 %0, %1, off"
               :: "v"(lds_byte_addr), "v"(gptr) : "memory");
}

// ---------- fp32 -> bf16 conversion ----------
__global__ __launch_bounds__(256) void cvt_bf16(const float* __restrict__ src,
                                                unsigned short* __restrict__ dst,
                                                int n) {
  int i = blockIdx.x * 256 + threadIdx.x;
  if (i < n) dst[i] = bf16_raw(src[i]);
}

// ---------- TN GEMM: out[m,n] = sum_k A[m,k] * W[n,k] + bias[n] ----------
// MODE 0: bf16 out, split-head [b,h,s,d]      (Q, K)
// MODE 1: bf16 out, transposed [b,h,d,s]      (V)
// MODE 2: fp32 out, row-major [m, n]          (final projection)
template <int MODE>
__global__ __launch_bounds__(256) void gemm_tn(const unsigned short* __restrict__ A,
                                               const unsigned short* __restrict__ Bw,
                                               const float* __restrict__ bias,
                                               void* __restrict__ outp) {
  const int wid  = (int)((blockIdx.x * blockDim.x + threadIdx.x) >> 5);
  const int lane = threadIdx.x & 31;
  const int nt64 = Dm / 64;                 // 16 column tiles of 64
  const int mt   = wid / nt64;              // 0..255
  const int nt   = wid % nt64;              // 0..15
  const int m0   = mt * 16, n0 = nt * 64;   // 64-wide tile == one head

  f32x8 acc[4] = {};
  const unsigned short* Arow = A  + (size_t)m0 * Dm;
  const unsigned short* Brow = Bw + (size_t)n0 * Dm;
  for (int k0 = 0; k0 < Dm; k0 += 32) {
    bf16x16 af = load_fragA(Arow + k0, Dm);
#pragma unroll
    for (int j = 0; j < 4; ++j) {
      bf16x16 bfr = load_fragB(Brow + (size_t)(j * 16) * Dm + k0, Dm);
      acc[j] = wmma_bf16(af, bfr, acc[j]);
    }
  }

  // C layout: lane<16 -> (m=r, n=lane); lane>=16 -> (m=r+8, n=lane-16)
  const int nl    = lane & 15;
  const int mhalf = (lane >> 4) * 8;
#pragma unroll
  for (int j = 0; j < 4; ++j) {
    const int n   = n0 + j * 16 + nl;
    const float bv = bias[n];
#pragma unroll
    for (int r = 0; r < 8; ++r) {
      const int m    = m0 + mhalf + r;
      const float val = acc[j][r] + bv;
      if (MODE == 2) {
        ((float*)outp)[(size_t)m * Dm + n] = val;
      } else {
        const int b = m >> 11, s = m & (Sq - 1);
        const int h = n >> 6,  d = n & 63;
        unsigned short* o = (unsigned short*)outp;
        if (MODE == 0)
          o[(((size_t)(b * Hh + h)) * Sq + s) * DEP + d] = bf16_raw(val);
        else
          o[(((size_t)(b * Hh + h)) * DEP + d) * Sq + s] = bf16_raw(val);
      }
    }
  }
}

// ---------- flash attention core: one wave = one (b,h) x 16 queries ----------
// Sᵀ = K·Qᵀ (queries in lanes -> 1 permlanex16 row reduction), Oᵀ = Vᵀ·Pᵀ with
// V pre-transposed [d,s]. K/V tiles double-buffered in LDS via async DMA.
// LDS held as u32x4[] and indexed with unsigned 16B-granule offsets so every
// fragment read is provably addrspace(3) -> ds_load_b128 (no FLAT fallback).
__global__ __launch_bounds__(256) void attn_fa(const unsigned short* __restrict__ Q,
                                               const unsigned short* __restrict__ Kx,
                                               const unsigned short* __restrict__ Vt,
                                               unsigned short* __restrict__ Obuf) {
  // 16B granules: per wave 2 buffers x (K 256 + V 256) = 1024 granules = 16KB
  __shared__ u32x4 smem4[8 * 1024];                       // 128KB total

  const int warp = threadIdx.x >> 5;
  const int lane = threadIdx.x & 31;
  const int wid  = (int)((blockIdx.x * blockDim.x + threadIdx.x) >> 5);
  const int qt   = wid & (Sq / 16 - 1);     // 0..127
  const int bh   = wid / (Sq / 16);         // 0..31
  const int q0   = qt * 16;
  const unsigned short* Qp = Q  + (size_t)bh * Sq * DEP;
  const unsigned short* Kp = Kx + (size_t)bh * Sq * DEP;
  const unsigned short* Vp = Vt + (size_t)bh * DEP * Sq;   // [d][s]

  const unsigned slab = (unsigned)warp * 1024u;            // wave's granule base
  auto kQ = [&](int buf) -> unsigned { return slab + (unsigned)buf * 512u; };
  auto vQ = [&](int buf) -> unsigned { return slab + (unsigned)buf * 512u + 256u; };

  // issue one K+V tile (4KB each) as 16 async b128 DMAs (32 lanes x 16B each)
  auto issue_tile = [&](int buf, int kt) {
    const unsigned kq = kQ(buf), vq = vQ(buf);
    const unsigned short* kg = Kp + (size_t)kt * DEP;      // contiguous 4KB
#pragma unroll
    for (int j = 0; j < 8; ++j)
      async_ld16((unsigned)(size_t)&smem4[kq + (unsigned)(j * 32 + lane)],
                 kg + j * 256 + lane * 8);
#pragma unroll
    for (int j = 0; j < 8; ++j) {                          // V: 64 rows x 64B
      const int idx = j * 32 + lane;                       // 0..255 chunks
      const int d = idx >> 2, c = idx & 3;
      async_ld16((unsigned)(size_t)&smem4[vq + (unsigned)(d * 4 + c)],
                 Vp + (size_t)d * Sq + kt + c * 8);
    }
  };

  // A-fragment straight from LDS granules: ldq = row stride in 16B units
  auto ldsFragA = [&](unsigned qbase, int ldq) -> bf16x16 {
    const unsigned idx = qbase + (unsigned)((lane & 15) * ldq) + (unsigned)(lane >> 4);
    FragU f;
    f.q[0] = smem4[idx];                                   // K = kb .. kb+7
    f.q[1] = smem4[idx + 2];                               // K = kb+16 .. kb+23
    return f.v;
  };

  // Q as B operand (columns = queries), K-dim = depth; loaded once from global
  const bf16x16 qf0 = load_fragB(Qp + (size_t)q0 * DEP +  0, DEP);
  const bf16x16 qf1 = load_fragB(Qp + (size_t)q0 * DEP + 32, DEP);

  float m_run = -1e30f, l_run = 0.f;       // exp2-domain running max / sum
  f32x8 o[4] = {};                         // Oᵀ: 4 depth tiles x 16 queries
  const float scl2 = 0.125f * 1.44269504088896340736f;  // (1/sqrt(64))*log2(e)

  issue_tile(0, 0);

  for (int it = 0; it < NKT; ++it) {
    const unsigned kc = kQ(it & 1);
    const unsigned vc = vQ(it & 1);
    if (it + 1 < NKT) {
      // prior ds reads of the other buffer must retire before DMA overwrites it
      asm volatile("s_wait_dscnt 0x0" ::: "memory");
      issue_tile((it + 1) & 1, (it + 1) * KT);
      asm volatile("s_wait_asynccnt 0x10" ::: "memory");   // tile `it` landed
    } else {
      asm volatile("s_wait_asynccnt 0x0" ::: "memory");
    }

    // K tile rows are 8 granules (128B); V tile rows are 4 granules (64B)
    bf16x16 ka0 = ldsFragA(kc,            8);   // keys 0..15,  d 0..31
    bf16x16 ka1 = ldsFragA(kc + 4,        8);   // keys 0..15,  d 32..63
    bf16x16 kb0 = ldsFragA(kc + 16 * 8,   8);   // keys 16..31, d 0..31
    bf16x16 kb1 = ldsFragA(kc + 16 * 8 + 4, 8); // keys 16..31, d 32..63
    f32x8 s0 = {}, s1 = {};
    s0 = wmma_bf16(ka0, qf0, s0);  s0 = wmma_bf16(ka1, qf1, s0);
    s1 = wmma_bf16(kb0, qf0, s1);  s1 = wmma_bf16(kb1, qf1, s1);

    // online softmax in exp2 domain; query = lane&15, lane holds 16 of 32 keys
    float mloc = -1e30f;
#pragma unroll
    for (int r = 0; r < 8; ++r) {
      s0[r] *= scl2;  s1[r] *= scl2;
      mloc = fmaxf(mloc, fmaxf(s0[r], s1[r]));
    }
    mloc = fmaxf(mloc, swap16f(mloc));
    const float m_new = fmaxf(m_run, mloc);
    const float alpha = __builtin_amdgcn_exp2f(m_run - m_new);
    float p0[8], p1[8], psum = 0.f;
#pragma unroll
    for (int r = 0; r < 8; ++r) {
      p0[r] = __builtin_amdgcn_exp2f(s0[r] - m_new);
      p1[r] = __builtin_amdgcn_exp2f(s1[r] - m_new);
      psum += p0[r] + p1[r];
    }
    psum += swap16f(psum);
    l_run = l_run * alpha + psum;
    m_run = m_new;
#pragma unroll
    for (int j = 0; j < 4; ++j)
#pragma unroll
      for (int r = 0; r < 8; ++r) o[j][r] *= alpha;

    // Repack Pᵀ (keys x queries) into B-fragment: lane<16 needs keys {0..7,16..23},
    // lane>=16 needs keys {8..15,24..31}: all local, no exchange.
    unsigned a4[4], b4[4];
#pragma unroll
    for (int j = 0; j < 4; ++j) {
      a4[j] = pack2(p0[2 * j], p0[2 * j + 1]);   // my s0 keys, pairs
      b4[j] = pack2(p1[2 * j], p1[2 * j + 1]);   // my s1 keys, pairs
    }
    FragU pf;
    const bool lohalf = (lane < 16);
#pragma unroll
    for (int j = 0; j < 4; ++j) {
      pf.dw[j]     = a4[j];                      // keys {0..7 | 8..15}
      pf.dw[4 + j] = b4[j];                      // keys {16..23 | 24..31}
    }

    // Oᵀ += Vᵀ · Pᵀ  (A = V-tile rows = depth, K-dim = 32 keys)
#pragma unroll
    for (int dt = 0; dt < 4; ++dt) {
      bf16x16 vf = ldsFragA(vc + (unsigned)(dt * 16 * 4), 4);
      o[dt] = wmma_bf16(vf, pf.v, o[dt]);
    }
  }

  // normalize and store merged-head bf16 [b*S+s][h*64+d]; 16B per tile per lane
  const float inv_l = 1.f / l_run;
  const int b = bh / Hh, h = bh % Hh;
  const int qloc  = lane & 15;
  const int dbase = (lane >> 4) * 8;
  unsigned short* rowp = Obuf + ((size_t)(b * Sq + q0 + qloc)) * Dm + h * DEP;
#pragma unroll
  for (int dt = 0; dt < 4; ++dt) {
    FragU st;
#pragma unroll
    for (int j = 0; j < 4; ++j)
      st.dw[j] = pack2(o[dt][2 * j] * inv_l, o[dt][2 * j + 1] * inv_l);
    *(u32x4*)(rowp + dt * 16 + dbase) = st.q[0];
  }
}

// ---------- host ----------
extern "C" void kernel_launch(void* const* d_in, const int* in_sizes, int n_in,
                              void* d_out, int out_size, void* d_ws, size_t ws_size,
                              hipStream_t stream) {
  const float* x  = (const float*)d_in[0];
  const float* Wq = (const float*)d_in[1];
  const float* bq = (const float*)d_in[2];
  const float* Wk = (const float*)d_in[3];
  const float* bk = (const float*)d_in[4];
  const float* Wv = (const float*)d_in[5];
  const float* bv = (const float*)d_in[6];
  const float* Wo = (const float*)d_in[7];
  const float* bo = (const float*)d_in[8];

  const size_t NX = (size_t)Mrows * Dm;     // 4096*1024
  const size_t NW = (size_t)Dm * Dm;        // 1024*1024
  char* ws = (char*)d_ws;
  unsigned short* xb   = (unsigned short*)(ws);
  unsigned short* wqb  = (unsigned short*)(ws + NX * 2);
  unsigned short* wkb  = (unsigned short*)(ws + NX * 2 + NW * 2);
  unsigned short* wvb  = (unsigned short*)(ws + NX * 2 + NW * 4);
  unsigned short* wob  = (unsigned short*)(ws + NX * 2 + NW * 6);
  unsigned short* qbuf = (unsigned short*)(ws + NX * 2 + NW * 8);
  unsigned short* kbuf = (unsigned short*)(ws + NX * 4 + NW * 8);
  unsigned short* vtbf = (unsigned short*)(ws + NX * 6 + NW * 8);
  unsigned short* attn = (unsigned short*)(ws + NX * 8 + NW * 8);

  // fp32 -> bf16
  cvt_bf16<<<(unsigned)(NX / 256), 256, 0, stream>>>(x,  xb,  (int)NX);
  cvt_bf16<<<(unsigned)(NW / 256), 256, 0, stream>>>(Wq, wqb, (int)NW);
  cvt_bf16<<<(unsigned)(NW / 256), 256, 0, stream>>>(Wk, wkb, (int)NW);
  cvt_bf16<<<(unsigned)(NW / 256), 256, 0, stream>>>(Wv, wvb, (int)NW);
  cvt_bf16<<<(unsigned)(NW / 256), 256, 0, stream>>>(Wo, wob, (int)NW);

  // (M/16)*(N/64) = 4096 waves -> 512 blocks of 8 waves
  gemm_tn<0><<<512, 256, 0, stream>>>(xb, wqb, bq, (void*)qbuf);
  gemm_tn<0><<<512, 256, 0, stream>>>(xb, wkb, bk, (void*)kbuf);
  gemm_tn<1><<<512, 256, 0, stream>>>(xb, wvb, bv, (void*)vtbf);

  // 32 (b,h) * 128 query tiles = 4096 waves -> 512 blocks
  attn_fa<<<512, 256, 0, stream>>>(qbuf, kbuf, vtbf, attn);

  gemm_tn<2><<<512, 256, 0, stream>>>(attn, wob, bo, d_out);
}
